// TaskBase_58102317580470
// MI455X (gfx1250) — matrix operand order repaired and checked by reading, not compiled, not run on hardware
//
#include <hip/hip_runtime.h>
#include <hip/hip_bf16.h>

#define B_ 64
#define T_ 2048
#define I_ 64
#define H_ 128
#define G_ 384   // 3*H
#define O_ 64

typedef __attribute__((ext_vector_type(16))) _Float16 v16h;
typedef __attribute__((ext_vector_type(8)))  float    v8f;
typedef __attribute__((ext_vector_type(4)))  float    f32x4;

union V16H { v16h h; f32x4 q[2]; };

// ---- WMMA wrapper: D = A(16x32 f16) * B(32x16 f16) + C(16x16 f32) ----
__device__ __forceinline__ v8f wmma_f16(v16h a, v16h b, v8f c) {
    return __builtin_amdgcn_wmma_f32_16x16x32_f16(
        /*neg_a=*/false, a, /*neg_b=*/false, b,
        /*c_mod=*/(short)0, c, /*reuse_a=*/false, /*reuse_b=*/false);
}

// A fragment (16x32): lane holds row m = lane%16.
// halves 0..7  -> k = k0 + 8*lg + j
// halves 8..15 -> k = k0 + 16 + 8*lg + j          (lg = lane/16)
__device__ __forceinline__ v16h load_a_row_f32(const float* __restrict__ row,
                                               int k0, int lg) {
    f32x4 q0 = *(const f32x4*)(row + k0 + 8 * lg);
    f32x4 q1 = *(const f32x4*)(row + k0 + 8 * lg + 4);
    f32x4 q2 = *(const f32x4*)(row + k0 + 16 + 8 * lg);
    f32x4 q3 = *(const f32x4*)(row + k0 + 16 + 8 * lg + 4);
    v16h a;
#pragma unroll
    for (int j = 0; j < 4; ++j) {
        a[j]      = (_Float16)q0[j];
        a[4 + j]  = (_Float16)q1[j];
        a[8 + j]  = (_Float16)q2[j];
        a[12 + j] = (_Float16)q3[j];
    }
    return a;
}

__device__ __forceinline__ v16h load_a_lds_f16(const _Float16* __restrict__ Hs,
                                               int stride, int m, int k0, int lg) {
    const _Float16* p0 = Hs + m * stride + k0 + 8 * lg;   // 16B aligned
    V16H u;
    u.q[0] = *(const f32x4*)(p0);
    u.q[1] = *(const f32x4*)(p0 + 16);
    return u.h;
}

// B fragment (32x16): lane holds col n = lane%16; half h<8 -> k = k0 + 8*lg + h, h>=8 -> k = k0 + 16 + 8*lg + (h-8).
// W stored in LDS as [n][k] (f16, padded stride) -> 32 contiguous halves.
__device__ __forceinline__ v16h load_b_lds_f16(const _Float16* __restrict__ W,
                                               int stride, int n, int k0, int lg) {
    const _Float16* p = W + n * stride + k0 + 8 * lg;    // 16B aligned
    V16H u;
    u.q[0] = *(const f32x4*)(p);
    u.q[1] = *(const f32x4*)(p + 16);
    return u.h;
}

// B fragment straight from a global f32 row (for register-resident weights):
// b[h] = W[n][k0 + 8*lg + h] (h<8), W[n][k0 + 16 + 8*lg + h-8] (h>=8), converted to f16.
__device__ __forceinline__ v16h load_b_row_f32(const float* __restrict__ row,
                                               int k0, int lg) {
    const float* p = row + k0 + 8 * lg;
    f32x4 q0 = *(const f32x4*)(p);
    f32x4 q1 = *(const f32x4*)(p + 4);
    f32x4 q2 = *(const f32x4*)(p + 16);
    f32x4 q3 = *(const f32x4*)(p + 20);
    v16h b;
#pragma unroll
    for (int j = 0; j < 4; ++j) {
        b[j]      = (_Float16)q0[j];
        b[4 + j]  = (_Float16)q1[j];
        b[8 + j]  = (_Float16)q2[j];
        b[12 + j] = (_Float16)q3[j];
    }
    return b;
}

__device__ __forceinline__ float sigmoidf_(float x) {
    return 1.0f / (1.0f + __expf(-x));
}

// ============================================================================
// Kernel 1: x_proj chunk = inputs[:, t0:t0+Tc, :] @ W_ih^T + b_ih  -> ws
// xp layout: [B][Tc][G], row m = b*Tc + tt.
// ============================================================================
__global__ __launch_bounds__(256) void k_xproj(
    const float* __restrict__ in,    // [B,T,I]
    const float* __restrict__ Wih,   // [G,I]
    const float* __restrict__ bih,   // [G]
    float* __restrict__ xp,          // [B,Tc,G]
    int t0, int Tc)
{
    __shared__ _Float16 Ws[G_ * (I_ + 8)];   // [n][k], stride 72 -> 55296 B
    const int tid  = threadIdx.x;
    const int lane = tid & 31, wv = tid >> 5;
    const int lg   = lane >> 4, nl = lane & 15;

    for (int idx = tid; idx < G_ * I_; idx += 256) {
        int n = idx >> 6, k = idx & 63;
        Ws[n * (I_ + 8) + k] = (_Float16)Wih[idx];
    }
    __syncthreads();

    const int m0   = (blockIdx.x * 8 + wv) * 16;      // row-tile base in [0, B*Tc)
    const int mrow = m0 + nl;                         // this lane's A row
    const int b    = mrow / Tc, tt = mrow % Tc;
    const float* arow = in + ((size_t)b * T_ + (size_t)(t0 + tt)) * I_;

    v16h a0 = load_a_row_f32(arow, 0, lg);
    v16h a1 = load_a_row_f32(arow, 32, lg);

    for (int n0 = 0; n0 < G_; n0 += 16) {
        v8f acc = {};
        v16h b0 = load_b_lds_f16(Ws, I_ + 8, n0 + nl, 0, lg);
        v16h b1 = load_b_lds_f16(Ws, I_ + 8, n0 + nl, 32, lg);
        acc = wmma_f16(a0, b0, acc);
        acc = wmma_f16(a1, b1, acc);
        float bias = bih[n0 + nl];
#pragma unroll
        for (int v = 0; v < 8; ++v) {
            int m = m0 + v + 8 * lg;
            xp[(size_t)m * G_ + n0 + nl] = acc[v] + bias;
        }
    }
}

// ============================================================================
// Kernel 2: sequential GRU scan. One workgroup, 32 waves (4 m-tiles x 8 col-
// tiles); each wave owns the (r,z,n) gate triple of its tile.
//  - fp32 h state lives in REGISTERS (ownership map is static across steps)
//  - only the f16 A-operand copy of h goes through LDS, double-buffered
//    => exactly ONE s_barrier per timestep
//  - all 12 W_hh B-fragments are hoisted into VGPRs before the loop
//    => per step: 8 ds_load_b128 + 12 v_wmma + gate VALU, nothing else on LDS
// ============================================================================
__global__ __launch_bounds__(1024) void k_scan(
    const float* __restrict__ xp,    // [B,Tc,G]
    const float* __restrict__ Whh,   // [G,H]
    const float* __restrict__ bhh,   // [G]
    float* __restrict__ lat,         // [B,T,H]
    float* __restrict__ hbuf,        // [B,H] persistent across chunks (ws)
    int t0, int Tc, int first)
{
    __shared__ _Float16 h16[2][B_ * (H_ + 8)];   // 2 x 17408 B

    const int tid  = threadIdx.x;
    const int lane = tid & 31, wv = tid >> 5;
    const int lg   = lane >> 4, nl = lane & 15;
    const int mt   = wv >> 3;                // 0..3  (rows  mt*16..+15)
    const int jt   = wv & 7;                 // 0..7  (cols  jt*16..+15)
    const int m_base = mt * 16;
    const int col    = jt * 16 + nl;         // 0..127

    // gate biases (scalar per lane)
    const float bR = bhh[col], bZ = bhh[H_ + col], bN = bhh[2 * H_ + col];

    // ---- hoist W_hh fragments: bf[gate][kf], b[h] = Whh[gate*H+col][k0+8lg+h] (h<8), [k0+16+8lg+h-8] (h>=8)
    v16h bf[3][4];
#pragma unroll
    for (int g = 0; g < 3; ++g) {
        const float* wrow = Whh + (size_t)(g * H_ + col) * H_;
#pragma unroll
        for (int kf = 0; kf < 4; ++kf)
            bf[g][kf] = load_b_row_f32(wrow, kf * 32, lg);
    }

    // ---- init h (registers + f16 LDS buffer 0) ----
    float hreg[8];
#pragma unroll
    for (int v = 0; v < 8; ++v) {
        int m = m_base + v + 8 * lg;
        float hv = first ? 0.0f : hbuf[m * H_ + col];
        hreg[v] = hv;
        h16[0][m * (H_ + 8) + col] = (_Float16)hv;
    }
    __syncthreads();

    for (int tt = 0; tt < Tc; ++tt) {
        const _Float16* hc = h16[tt & 1];
        _Float16*       hn = h16[(tt + 1) & 1];

        // x_proj loads issued first so global latency overlaps the WMMAs
        float xr[8], xz[8], xn_[8];
#pragma unroll
        for (int v = 0; v < 8; ++v) {
            int m = m_base + v + 8 * lg;
            const float* xrow = xp + ((size_t)m * Tc + tt) * G_;
            xr[v]  = xrow[col];
            xz[v]  = xrow[H_ + col];
            xn_[v] = xrow[2 * H_ + col];
        }

        // ---- hg = h @ W_hh^T for this wave's (r,z,n) tiles ----
        v8f aR = {}, aZ = {}, aN = {};
#pragma unroll
        for (int kf = 0; kf < 4; ++kf) {
            v16h a = load_a_lds_f16(hc, H_ + 8, m_base + nl, kf * 32, lg);
            aR = wmma_f16(a, bf[0][kf], aR);
            aZ = wmma_f16(a, bf[1][kf], aZ);
            aN = wmma_f16(a, bf[2][kf], aN);
        }

        // ---- fused gates + state update; h_old comes from registers ----
#pragma unroll
        for (int v = 0; v < 8; ++v) {
            int m = m_base + v + 8 * lg;
            float r  = sigmoidf_(xr[v] + aR[v] + bR);
            float z  = sigmoidf_(xz[v] + aZ[v] + bZ);
            float nn = tanhf(xn_[v] + r * (aN[v] + bN));
            float hv = (1.0f - z) * nn + z * hreg[v];
            hreg[v] = hv;
            hn[m * (H_ + 8) + col] = (_Float16)hv;     // other buffer: no race
            lat[((size_t)m * T_ + (size_t)(t0 + tt)) * H_ + col] = hv;
        }

        // prefetch next timestep's x_proj row (global_prefetch_b8)
        if (tt + 1 < Tc) {
            const float* nx = xp + ((size_t)(m_base + nl) * Tc + (tt + 1)) * G_;
            __builtin_prefetch(nx + col, 0, 1);
        }

        __syncthreads();   // publish h16[next] before anyone reads it
    }

    // persist h for next chunk
#pragma unroll
    for (int v = 0; v < 8; ++v) {
        int m = m_base + v + 8 * lg;
        hbuf[m * H_ + col] = hreg[v];
    }
}

// ============================================================================
// Kernel 3: output = latents @ W_out^T + b_out
// ============================================================================
__global__ __launch_bounds__(256) void k_out(
    const float* __restrict__ lat,   // [B,T,H]
    const float* __restrict__ Wout,  // [O,H]
    const float* __restrict__ bout,  // [O]
    float* __restrict__ out)         // [B,T,O]
{
    __shared__ _Float16 Ws[O_ * (H_ + 8)];   // 17408 B
    const int tid  = threadIdx.x;
    const int lane = tid & 31, wv = tid >> 5;
    const int lg   = lane >> 4, nl = lane & 15;

    for (int idx = tid; idx < O_ * H_; idx += 256) {
        int n = idx >> 7, k = idx & 127;
        Ws[n * (H_ + 8) + k] = (_Float16)Wout[idx];
    }
    __syncthreads();

    const size_t m0 = ((size_t)blockIdx.x * 8 + wv) * 16;
    const float* arow = lat + (m0 + nl) * (size_t)H_;

    v16h a[4];
#pragma unroll
    for (int kf = 0; kf < 4; ++kf) a[kf] = load_a_row_f32(arow, kf * 32, lg);

    for (int n0 = 0; n0 < O_; n0 += 16) {
        v8f acc = {};
#pragma unroll
        for (int kf = 0; kf < 4; ++kf) {
            v16h b = load_b_lds_f16(Ws, H_ + 8, n0 + nl, kf * 32, lg);
            acc = wmma_f16(a[kf], b, acc);
        }
        float bias = bout[n0 + nl];
#pragma unroll
        for (int v = 0; v < 8; ++v) {
            size_t m = m0 + v + 8 * lg;
            out[m * O_ + n0 + nl] = acc[v] + bias;
        }
    }
}

// ============================================================================
extern "C" void kernel_launch(void* const* d_in, const int* in_sizes, int n_in,
                              void* d_out, int out_size, void* d_ws, size_t ws_size,
                              hipStream_t stream)
{
    const float* in   = (const float*)d_in[0];
    const float* Wih  = (const float*)d_in[1];
    const float* Whh  = (const float*)d_in[2];
    const float* bih  = (const float*)d_in[3];
    const float* bhh  = (const float*)d_in[4];
    const float* Wout = (const float*)d_in[5];
    const float* bout = (const float*)d_in[6];

    float* out = (float*)d_out;                       // [B,T,O]
    float* lat = out + (size_t)B_ * T_ * O_;          // [B,T,H]

    // Pick time-chunk Tc so x_proj staging + h buffer fit in workspace.
    int Tc = T_;
    while (Tc > 16 &&
           ((size_t)B_ * Tc * G_ * sizeof(float) + (size_t)B_ * H_ * sizeof(float)) > ws_size)
        Tc >>= 1;
    float* xp   = (float*)d_ws;                       // [B,Tc,G]
    float* hbuf = xp + (size_t)B_ * Tc * G_;          // [B,H]

    const int nchunks = T_ / Tc;
    for (int c = 0; c < nchunks; ++c) {
        const int t0 = c * Tc;
        k_xproj<<<dim3((B_ * Tc) / 128), dim3(256), 0, stream>>>(in, Wih, bih, xp, t0, Tc);
        k_scan <<<dim3(1), dim3(1024), 0, stream>>>(xp, Whh, bhh, lat, hbuf, t0, Tc, c == 0 ? 1 : 0);
    }
    k_out<<<dim3((B_ * T_) / 128), dim3(256), 0, stream>>>(lat, Wout, bout, out);
}
